// TriangleAttentionStartingNode_7456063225920
// MI455X (gfx1250) — matrix operand order repaired and checked by reading, compiled, not run on hardware
//
#include <hip/hip_runtime.h>
#include <hip/hip_bf16.h>
#include <math.h>

typedef __attribute__((ext_vector_type(16))) __bf16 v16bf;
typedef __attribute__((ext_vector_type(8)))  __bf16 v8bf;
typedef __attribute__((ext_vector_type(8)))  float  v8f;
typedef __attribute__((ext_vector_type(4)))  float  v4f;

union AB16 { v16bf v; v8bf h[2]; __bf16 e[16]; };

__device__ __forceinline__ __bf16 to_bf(float x) { return (__bf16)x; }

// ---------------------------------------------------------------------------
// Weight packing: per 128x128 matrix (K rows x N cols, row-major source),
// fragment f = nt*4 + kt (nt: N-tile of 16, kt: K-tile of 32).
// Within fragment: lane*16 + e, element = W[K*128+N] with
//   K = kt*32 + (lane>>4)*8 + (e&7) + (e>>3)*16,  N = nt*16 + (lane&15)
// -> GEMM waves read each B fragment as 32 contiguous bytes per lane.
// ---------------------------------------------------------------------------
__global__ __launch_bounds__(256) void pack_weights(
    const float* __restrict__ Wq, const float* __restrict__ Wk,
    const float* __restrict__ Wv, const float* __restrict__ Wg,
    const float* __restrict__ Wo, __bf16* __restrict__ dst)
{
  int id  = blockIdx.x * 256 + threadIdx.x;   // 0 .. 5*16384-1
  int mat = id >> 14;
  int r   = id & 16383;
  int f = r >> 9, lane = (r >> 4) & 31, e = r & 15;
  int kt = f & 3, nt = f >> 2;
  int K    = kt * 32 + (lane >> 4) * 8 + (e & 7) + (e >> 3) * 16;
  int Ncol = nt * 16 + (lane & 15);
  const float* W = (mat == 0) ? Wq : (mat == 1) ? Wk : (mat == 2) ? Wv
                 : (mat == 3) ? Wg : Wo;
  dst[id] = to_bf(W[K * 128 + Ncol]);
}

// ---------------------------------------------------------------------------
// Fused LayerNorm + {q,k,v,g} projections + pair bias.
// 16 rows per block, 8 waves; each wave produces 4 16x16 output tiles.
// q,k row-major bf16 [65536,128]; v transposed vT[i][hc][j]; g pre-sigmoided.
// ---------------------------------------------------------------------------
__global__ __launch_bounds__(256) void ln_proj(
    const float* __restrict__ z, const float* __restrict__ gamma,
    const float* __restrict__ beta, const float* __restrict__ Wb,
    const __bf16* __restrict__ wpack,
    __bf16* __restrict__ qo, __bf16* __restrict__ ko,
    __bf16* __restrict__ vT, __bf16* __restrict__ sg,
    float* __restrict__ biasO)
{
  __shared__ float  zraw[16 * 128];
  __shared__ __bf16 zbf[16 * 128];
  __shared__ float  mu[16], rstd[16];
  int t = threadIdx.x;
  int rowbase = blockIdx.x * 16;

  const float* src = z + (size_t)rowbase * 128;
  for (int u = 0; u < 2; ++u) {
    int fidx = (u * 256 + t) * 4;
    *(v4f*)&zraw[fidx] = *(const v4f*)&src[fidx];
  }
  __syncthreads();

  {  // LN stats: 16 lanes per row, shfl_xor tree within 16-lane halves
    int row = t >> 4, seg = t & 15;
    float s1 = 0.f, s2 = 0.f;
#pragma unroll
    for (int e = 0; e < 8; ++e) {
      float x = zraw[row * 128 + seg * 8 + e];
      s1 += x; s2 += x * x;
    }
#pragma unroll
    for (int m = 1; m < 16; m <<= 1) {
      s1 += __shfl_xor(s1, m, 32);
      s2 += __shfl_xor(s2, m, 32);
    }
    float mean = s1 * (1.f / 128.f);
    float var  = s2 * (1.f / 128.f) - mean * mean;
    if (seg == 0) { mu[row] = mean; rstd[row] = rsqrtf(var + 1e-5f); }
  }
  __syncthreads();

  // normalize -> bf16 activations in LDS
  for (int u = 0; u < 2; ++u) {
    int fidx = (u * 256 + t) * 4;
    int row = fidx >> 7;
    float m = mu[row], rs = rstd[row];
#pragma unroll
    for (int e2 = 0; e2 < 4; ++e2) {
      int c = (fidx + e2) & 127;
      float xv = (zraw[fidx + e2] - m) * rs * gamma[c] + beta[c];
      zbf[fidx + e2] = to_bf(xv);
    }
  }
  // pair bias = zln @ Wb  (64 threads, f32 VALU; tiny)
  if (t < 64) {
    int row = t >> 2, hh = t & 3;
    float m = mu[row], rs = rstd[row], acc = 0.f;
    for (int c = 0; c < 128; ++c) {
      float xv = (zraw[row * 128 + c] - m) * rs * gamma[c] + beta[c];
      acc += xv * Wb[c * 4 + hh];
    }
    biasO[(size_t)(rowbase + row) * 4 + hh] = acc;
  }
  __syncthreads();

  int w = t >> 5, lane = t & 31;
  int half = lane >> 4, col = lane & 15, mrow = lane & 15;
  v8f acc0 = {}, acc1 = {}, acc2 = {}, acc3 = {};
#pragma unroll
  for (int kt = 0; kt < 4; ++kt) {
    AB16 a;
    int abase = half * 8;
    const __bf16* ap = &zbf[mrow * 128 + kt * 32 + abase];
    a.h[0] = *(const v8bf*)ap;
    a.h[1] = *(const v8bf*)(ap + 16);
#pragma unroll
    for (int q4 = 0; q4 < 4; ++q4) {
      int ntg = w * 4 + q4;
      int mat = ntg >> 3, nt = ntg & 7;
      const __bf16* bp = wpack + mat * 16384 + (nt * 4 + kt) * 512 + lane * 16;
      AB16 b;
      b.h[0] = *(const v8bf*)bp;
      b.h[1] = *(const v8bf*)(bp + 8);
      v8f c = (q4 == 0) ? acc0 : (q4 == 1) ? acc1 : (q4 == 2) ? acc2 : acc3;
      c = __builtin_amdgcn_wmma_f32_16x16x32_bf16(false, a.v, false, b.v,
                                                  (short)0, c, false, false);
      if (q4 == 0) acc0 = c; else if (q4 == 1) acc1 = c;
      else if (q4 == 2) acc2 = c; else acc3 = c;
    }
  }
#pragma unroll
  for (int q4 = 0; q4 < 4; ++q4) {
    int ntg = w * 4 + q4;
    int mat = ntg >> 3, nt = ntg & 7;
    v8f c = (q4 == 0) ? acc0 : (q4 == 1) ? acc1 : (q4 == 2) ? acc2 : acc3;
#pragma unroll
    for (int r = 0; r < 8; ++r) {
      int m = r + half * 8;
      int grow = rowbase + m;
      int ncol = nt * 16 + col;
      float v = c[r];
      if (mat == 0)      qo[(size_t)grow * 128 + ncol] = to_bf(v);
      else if (mat == 1) ko[(size_t)grow * 128 + ncol] = to_bf(v);
      else if (mat == 2) {
        int i = grow >> 8, j = grow & 255;
        vT[((size_t)i * 128 + ncol) * 256 + j] = to_bf(v);
      } else {
        sg[(size_t)grow * 128 + ncol] = to_bf(1.f / (1.f + __expf(-v)));
      }
    }
  }
}

// ---------------------------------------------------------------------------
// Attention per (i, h, j-half): each wave owns 16 j rows; scores via WMMA
// into per-wave LDS (16KB), bias + softmax over k, then P@V via WMMA; gate.
// ---------------------------------------------------------------------------
__global__ __launch_bounds__(256) void attn(
    const __bf16* __restrict__ qm, const __bf16* __restrict__ km,
    const __bf16* __restrict__ vT, const __bf16* __restrict__ sg,
    const float* __restrict__ bias, __bf16* __restrict__ og)
{
  extern __shared__ float smem[];
  int jblk = blockIdx.x, h = blockIdx.y, i = blockIdx.z;
  int t = threadIdx.x, w = t >> 5, lane = t & 31;
  int half = lane >> 4, col = lane & 15;
  float* u = smem + w * (16 * 256);
  int j0 = jblk * 128 + w * 16;

  AB16 qa;  // Q fragment (A: 16 j x 32 d)
  {
    int j = j0 + (lane & 15);
    int abase = half * 16;
    const __bf16* qp = qm + ((size_t)(i * 256 + j) * 128 + h * 32 + abase);
    qa.h[0] = *(const v8bf*)qp;
    qa.h[1] = *(const v8bf*)(qp + 8);
  }
  const float scale = 0.17677669529663687f;  // 1/sqrt(32)

  for (int kt = 0; kt < 16; ++kt) {          // S = Q K^T
    int k0 = kt * 16;
    AB16 kb;
    const __bf16* kp = km + ((size_t)(i * 256 + k0 + col) * 128 + h * 32 + half * 16);
    kb.h[0] = *(const v8bf*)kp;
    kb.h[1] = *(const v8bf*)(kp + 8);
    v8f zero = {};
    v8f s = __builtin_amdgcn_wmma_f32_16x16x32_bf16(false, qa.v, false, kb.v,
                                                    (short)0, zero, false, false);
#pragma unroll
    for (int r = 0; r < 8; ++r) {
      int m = r + half * 8;
      u[m * 256 + k0 + col] = s[r] * scale;
    }
  }
  __syncthreads();

  // bias + softmax: lane handles row (lane&15), k-half (lane>>4)
  int jr = lane & 15;
  int kbase = half * 128;
  float mx = -1e30f;
  for (int tt = 0; tt < 128; ++tt) {
    int kk = kbase + tt;
    float x = u[jr * 256 + kk] + bias[((size_t)(j0 + jr) * 256 + kk) * 4 + h];
    u[jr * 256 + kk] = x;
    mx = fmaxf(mx, x);
  }
  mx = fmaxf(mx, __shfl_xor(mx, 16, 32));
  float sum = 0.f;
  for (int tt = 0; tt < 128; ++tt) {
    int kk = kbase + tt;
    float e = __expf(u[jr * 256 + kk] - mx);
    u[jr * 256 + kk] = e;
    sum += e;
  }
  sum += __shfl_xor(sum, 16, 32);
  float rs = 1.f / sum;
  __syncthreads();

  // O = P @ V  (d = 32 -> two 16-col tiles); A lane row == jr so rs is local
  v8f o0 = {}, o1 = {};
  for (int kt = 0; kt < 8; ++kt) {
    int k0 = kt * 32;
    AB16 pa;
    int abase = half * 16;
    const float* up = &u[jr * 256 + k0 + abase];
#pragma unroll
    for (int e = 0; e < 8; ++e) pa.e[e] = to_bf(up[e] * rs);
#pragma unroll
    for (int e = 0; e < 8; ++e) pa.e[8 + e] = to_bf(up[8 + e] * rs);
    const __bf16* vp0 = vT + ((size_t)(i * 128 + h * 32 + col) * 256 + k0 + half * 16);
    const __bf16* vp1 = vT + ((size_t)(i * 128 + h * 32 + 16 + col) * 256 + k0 + half * 16);
    AB16 b0, b1;
    b0.h[0] = *(const v8bf*)vp0; b0.h[1] = *(const v8bf*)(vp0 + 8);
    b1.h[0] = *(const v8bf*)vp1; b1.h[1] = *(const v8bf*)(vp1 + 8);
    o0 = __builtin_amdgcn_wmma_f32_16x16x32_bf16(false, pa.v, false, b0.v,
                                                 (short)0, o0, false, false);
    o1 = __builtin_amdgcn_wmma_f32_16x16x32_bf16(false, pa.v, false, b1.v,
                                                 (short)0, o1, false, false);
  }
#pragma unroll
  for (int r = 0; r < 8; ++r) {
    int m = r + half * 8;
    size_t rowoff = (size_t)(i * 256 + j0 + m) * 128 + h * 32;
    float g0 = (float)sg[rowoff + col];
    float g1 = (float)sg[rowoff + 16 + col];
    og[rowoff + col]      = to_bf(o0[r] * g0);
    og[rowoff + 16 + col] = to_bf(o1[r] * g1);
  }
}

// ---------------------------------------------------------------------------
// out = og @ Wout  (f32 result)
// ---------------------------------------------------------------------------
__global__ __launch_bounds__(256) void out_proj(
    const __bf16* __restrict__ og, const __bf16* __restrict__ wo,
    float* __restrict__ out)
{
  __shared__ __bf16 obf[16 * 128];
  int t = threadIdx.x;
  int rowbase = blockIdx.x * 16;
  {
    const __bf16* src = og + (size_t)rowbase * 128;
    *(v8bf*)&obf[t * 8] = *(const v8bf*)&src[t * 8];
  }
  __syncthreads();
  int w = t >> 5, lane = t & 31, half = lane >> 4, col = lane & 15;
  int mrow = lane & 15;
  v8f acc = {};
#pragma unroll
  for (int kt = 0; kt < 4; ++kt) {
    AB16 a;
    const __bf16* ap = &obf[mrow * 128 + kt * 32 + half * 8];
    a.h[0] = *(const v8bf*)ap;
    a.h[1] = *(const v8bf*)(ap + 16);
    const __bf16* bp = wo + (w * 4 + kt) * 512 + lane * 16;
    AB16 b;
    b.h[0] = *(const v8bf*)bp;
    b.h[1] = *(const v8bf*)(bp + 8);
    acc = __builtin_amdgcn_wmma_f32_16x16x32_bf16(false, a.v, false, b.v,
                                                  (short)0, acc, false, false);
  }
#pragma unroll
  for (int r = 0; r < 8; ++r) {
    int m = r + half * 8;
    out[(size_t)(rowbase + m) * 128 + w * 16 + col] = acc[r];
  }
}

extern "C" void kernel_launch(void* const* d_in, const int* in_sizes, int n_in,
                              void* d_out, int out_size, void* d_ws, size_t ws_size,
                              hipStream_t stream)
{
  const float* z     = (const float*)d_in[0];
  const float* gamma = (const float*)d_in[1];
  const float* beta  = (const float*)d_in[2];
  const float* Wq    = (const float*)d_in[3];
  const float* Wk    = (const float*)d_in[4];
  const float* Wv    = (const float*)d_in[5];
  const float* Wb    = (const float*)d_in[6];
  const float* Wg    = (const float*)d_in[7];
  const float* Wo    = (const float*)d_in[8];
  float* out = (float*)d_out;

  char* ws = (char*)d_ws;
  size_t off = 0;
  auto carve = [&](size_t bytes) -> char* {
    char* p = ws + off;
    off = (off + bytes + 255) & ~(size_t)255;
    return p;
  };
  __bf16* wpack = (__bf16*)carve((size_t)5 * 16384 * 2);
  __bf16* qm    = (__bf16*)carve((size_t)65536 * 128 * 2);
  __bf16* km    = (__bf16*)carve((size_t)65536 * 128 * 2);
  __bf16* vT    = (__bf16*)carve((size_t)65536 * 128 * 2);
  __bf16* sg    = (__bf16*)carve((size_t)65536 * 128 * 2);
  __bf16* og    = (__bf16*)carve((size_t)65536 * 128 * 2);
  float*  bias  = (float*)carve((size_t)65536 * 4 * 4);

  pack_weights<<<320, 256, 0, stream>>>(Wq, Wk, Wv, Wg, Wo, wpack);
  ln_proj<<<4096, 256, 0, stream>>>(z, gamma, beta, Wb, wpack, qm, km, vT, sg, bias);
  dim3 g2(2, 4, 256);
  attn<<<g2, 256, 131072, stream>>>(qm, km, vT, sg, bias, og);
  out_proj<<<4096, 256, 0, stream>>>(og, wpack + 4 * 16384, out);
}
